// trajectory2seq_53919019434383
// MI455X (gfx1250) — compile-verified, matrix operand order checked by reading, not run on hardware
//
#include <hip/hip_runtime.h>
#include <hip/hip_bf16.h>

#define H   1024
#define G   3072          // 3*H
#define Bv  50            // valid batch
#define Bp  64            // padded batch (4 M-tiles)
#define Sn  457
#define Tn  32
#define Vn  27
#define NBLK 64
#define NTHR 256

#define OHID  (Bv*Tn*Vn)                 // 43200
#define OATTN (OHID + 2*Bv*H)            // 145600

typedef __attribute__((ext_vector_type(16))) __bf16        v16bf;
typedef __attribute__((ext_vector_type(8)))  float         v8f;
typedef __attribute__((ext_vector_type(8)))  unsigned int  v8u;

union FragU { v8u u; v16bf b; };

__device__ __forceinline__ unsigned short f2bf(float f) {
  unsigned int u = __float_as_uint(f);
  u += 0x7fffu + ((u >> 16) & 1u);      // round-to-nearest-even
  return (unsigned short)(u >> 16);
}
__device__ __forceinline__ float bf2f(unsigned short h) {
  return __uint_as_float(((unsigned int)h) << 16);
}
__device__ __forceinline__ float sigmoidf_(float x) { return 1.0f / (1.0f + __expf(-x)); }

// ---- WMMA fragment loaders (bf16, 16x16x32) ------------------------------
// A (MxK) row-major: lane l holds row (l&15); K pairs striped per ISA layout.
// B operand: weights stored as W[n][k] (i.e. B^T row-major); same loader with
// "row"=n — symmetric K striping.
__device__ __forceinline__ v16bf load_frag(const unsigned short* __restrict__ base,
                                           int ld, int row0, int k0) {
  int lane = threadIdx.x & 31;
  const unsigned int* q = (const unsigned int*)
      (base + (size_t)(row0 + (lane & 15)) * ld + (k0 + ((lane >> 4) << 3)));
  FragU f;
#pragma unroll
  for (int v = 0; v < 8; ++v) f.u[v] = q[(v & 3) + ((v >> 2) << 3)];
  return f.b;
}
// gather variant: row -> emb[token[row]]
__device__ __forceinline__ v16bf load_frag_gather(const unsigned short* __restrict__ emb,
                                                  const int* __restrict__ tok,
                                                  int row0, int k0) {
  int lane = threadIdx.x & 31;
  int t = tok[row0 + (lane & 15)];
  const unsigned int* q = (const unsigned int*)
      (emb + (size_t)t * H + (k0 + ((lane >> 4) << 3)));
  FragU f;
#pragma unroll
  for (int v = 0; v < 8; ++v) f.u[v] = q[(v & 3) + ((v >> 2) << 3)];
  return f.b;
}

#define WMMA_BF16(a, b, c) \
  __builtin_amdgcn_wmma_f32_16x16x32_bf16(false, (a), false, (b), (short)0, (c), false, false)

// ---- device-scope grid barrier (64 resident blocks) ----------------------
__device__ __forceinline__ void grid_barrier(int* cnt, int* gen) {
  __syncthreads();
  if (threadIdx.x == 0) {
    __threadfence();
    int g = __hip_atomic_load(gen, __ATOMIC_RELAXED, __HIP_MEMORY_SCOPE_AGENT);
    int t = __hip_atomic_fetch_add(cnt, 1, __ATOMIC_ACQ_REL, __HIP_MEMORY_SCOPE_AGENT);
    if (t == NBLK - 1) {
      __hip_atomic_store(cnt, 0, __ATOMIC_RELAXED, __HIP_MEMORY_SCOPE_AGENT);
      __hip_atomic_fetch_add(gen, 1, __ATOMIC_RELEASE, __HIP_MEMORY_SCOPE_AGENT);
    } else {
      while (__hip_atomic_load(gen, __ATOMIC_ACQUIRE, __HIP_MEMORY_SCOPE_AGENT) == g)
        __builtin_amdgcn_s_sleep(1);
    }
    __threadfence();
  }
  __syncthreads();
}

// ---- fused GRU step: waves 0-3 do h@Whh^T, waves 4-7 do x@Wih^T ----------
__device__ __forceinline__ void gru_fused_step(
    const unsigned short* __restrict__ Ax,     // input activations (64xH bf16) or emb table
    const int* __restrict__ tok,               // non-null -> gather rows from emb
    const unsigned short* __restrict__ hread,  // previous hidden (bf16 ping)
    const unsigned short* __restrict__ Wih,    // (3H x H) bf16
    const unsigned short* __restrict__ Whh,    // (3H x H) bf16
    const float* __restrict__ bih, const float* __restrict__ bhh,
    float* __restrict__ hf,                    // fp32 hidden master (64xH)
    unsigned short* __restrict__ hwrite,       // bf16 pong
    unsigned short* __restrict__ extra, int extra_ld,  // optional extra bf16 store
    float (*lds_gi)[Bp][16]) {
  const int j = blockIdx.x;
  const int w = threadIdx.x >> 5;
  const int lane = threadIdx.x & 31;
  const int coll = lane & 15;
  const int roff = (lane >> 4) << 3;
  v8f aR = {}, aZ = {}, aN = {};
  if (w < 4) {                                   // recurrent GEMM, 3 gate tiles
    const int m = w * 16;
    for (int k0 = 0; k0 < H; k0 += 32) {
      v16bf a = load_frag(hread, H, m, k0);
      aR = WMMA_BF16(a, load_frag(Whh, H, 0 * H + j * 16, k0), aR);
      aZ = WMMA_BF16(a, load_frag(Whh, H, 1 * H + j * 16, k0), aZ);
      aN = WMMA_BF16(a, load_frag(Whh, H, 2 * H + j * 16, k0), aN);
    }
  } else {                                       // input GEMM, published via LDS
    const int m = (w - 4) * 16;
    for (int k0 = 0; k0 < H; k0 += 32) {
      v16bf a = tok ? load_frag_gather(Ax, tok, m, k0) : load_frag(Ax, H, m, k0);
      aR = WMMA_BF16(a, load_frag(Wih, H, 0 * H + j * 16, k0), aR);
      aZ = WMMA_BF16(a, load_frag(Wih, H, 1 * H + j * 16, k0), aZ);
      aN = WMMA_BF16(a, load_frag(Wih, H, 2 * H + j * 16, k0), aN);
    }
    const int n = j * 16 + coll;
#pragma unroll
    for (int r = 0; r < 8; ++r) {
      int row = m + roff + r;
      lds_gi[0][row][coll] = aR[r] + bih[0 * H + n];
      lds_gi[1][row][coll] = aZ[r] + bih[1 * H + n];
      lds_gi[2][row][coll] = aN[r] + bih[2 * H + n];
    }
  }
  __syncthreads();
  if (w < 4) {                                   // GRU nonlinearity (local)
    const int n = j * 16 + coll;
    const float bR = bhh[0 * H + n], bZ = bhh[1 * H + n], bN = bhh[2 * H + n];
#pragma unroll
    for (int r = 0; r < 8; ++r) {
      int row = w * 16 + roff + r;
      float gr = sigmoidf_(lds_gi[0][row][coll] + aR[r] + bR);
      float gz = sigmoidf_(lds_gi[1][row][coll] + aZ[r] + bZ);
      float gn = tanhf(lds_gi[2][row][coll] + gr * (aN[r] + bN));
      float hp = hf[(size_t)row * H + n];
      float hv = (1.0f - gz) * gn + gz * hp;
      hf[(size_t)row * H + n] = hv;
      unsigned short hb = f2bf(hv);
      hwrite[(size_t)row * H + n] = hb;
      if (extra) extra[(size_t)row * extra_ld + n] = hb;
    }
  }
}

// ---- encoder layer 0: K_in = 3 (scalar), so split-K the recurrent GEMM ---
// Waves 0-3: gh over K[0,512). Waves 4-7: gh over K[512,1024) + scalar gi.
// Partials exchanged through LDS; halves the serial WMMA chain per step.
__global__ void __launch_bounds__(NTHR) enc_l0_kernel(
    const float* __restrict__ x, const float* __restrict__ Wih0,
    const float* __restrict__ bih0, const float* __restrict__ bhh0,
    const unsigned short* __restrict__ Whh0_bf,
    unsigned short* __restrict__ h0_bf, float* __restrict__ h0_f,
    unsigned short* __restrict__ seq0_bf, int* cnt, int* gen) {
  __shared__ float lds_gi[3][Bp][16];
  __shared__ float lds_gh[3][Bp][16];
  const int j = blockIdx.x, w = threadIdx.x >> 5, lane = threadIdx.x & 31;
  const int coll = lane & 15, roff = (lane >> 4) << 3;
  const int m = (w & 3) * 16;
  const int kbase = (w < 4) ? 0 : 512;
  const int n = j * 16 + coll;
  for (int t = 0; t < Sn; ++t) {
    const int rp = t & 1, wp = rp ^ 1;
    const unsigned short* hb = h0_bf + (size_t)rp * Bp * H;
    v8f aR = {}, aZ = {}, aN = {};
    for (int k0 = kbase; k0 < kbase + 512; k0 += 32) {
      v16bf a = load_frag(hb, H, m, k0);
      aR = WMMA_BF16(a, load_frag(Whh0_bf, H, 0 * H + j * 16, k0), aR);
      aZ = WMMA_BF16(a, load_frag(Whh0_bf, H, 1 * H + j * 16, k0), aZ);
      aN = WMMA_BF16(a, load_frag(Whh0_bf, H, 2 * H + j * 16, k0), aN);
    }
    if (w >= 4) {
      // publish gh partial (upper K half)
#pragma unroll
      for (int r = 0; r < 8; ++r) {
        int row = m + roff + r;
        lds_gh[0][row][coll] = aR[r];
        lds_gh[1][row][coll] = aZ[r];
        lds_gh[2][row][coll] = aN[r];
      }
      // scalar input-GEMM (K=3) + bias
      const float ps = (float)t;
#pragma unroll
      for (int r = 0; r < 8; ++r) {
        int b = m + roff + r;
        float x0 = (b < Bv) ? x[(size_t)b * 2 * Sn + t] : 0.0f;
        float x1 = (b < Bv) ? x[(size_t)(b * 2 + 1) * Sn + t] : 0.0f;
#pragma unroll
        for (int g = 0; g < 3; ++g) {
          int nn = g * H + n;
          lds_gi[g][b][coll] =
              Wih0[nn * 3 + 0] * x0 + Wih0[nn * 3 + 1] * x1 + Wih0[nn * 3 + 2] * ps + bih0[nn];
        }
      }
    }
    __syncthreads();
    if (w < 4) {
      const float bR = bhh0[0 * H + n], bZ = bhh0[1 * H + n], bN = bhh0[2 * H + n];
#pragma unroll
      for (int r = 0; r < 8; ++r) {
        int row = m + roff + r;
        float gr = sigmoidf_(lds_gi[0][row][coll] + aR[r] + lds_gh[0][row][coll] + bR);
        float gz = sigmoidf_(lds_gi[1][row][coll] + aZ[r] + lds_gh[1][row][coll] + bZ);
        float gn = tanhf(lds_gi[2][row][coll] +
                         gr * (aN[r] + lds_gh[2][row][coll] + bN));
        float hp = h0_f[(size_t)row * H + n];
        float hv = (1.0f - gz) * gn + gz * hp;
        h0_f[(size_t)row * H + n] = hv;
        unsigned short hbv = f2bf(hv);
        h0_bf[(size_t)wp * Bp * H + (size_t)row * H + n] = hbv;
        seq0_bf[(size_t)t * Bp * H + (size_t)row * H + n] = hbv;
      }
    }
    grid_barrier(cnt, gen);
  }
}

// ---- encoder layer 1 (both GEMMs via WMMA) -------------------------------
__global__ void __launch_bounds__(NTHR) enc_l1_kernel(
    const unsigned short* __restrict__ seq0_bf,
    const unsigned short* __restrict__ Wih1_bf, const unsigned short* __restrict__ Whh1_bf,
    const float* __restrict__ bih1, const float* __restrict__ bhh1,
    unsigned short* __restrict__ h1_bf, float* __restrict__ h1_f,
    unsigned short* __restrict__ enc_bf, int* cnt, int* gen) {
  __shared__ float lds_gi[3][Bp][16];
  for (int t = 0; t < Sn; ++t) {
    const int rp = t & 1, wp = rp ^ 1;
    gru_fused_step(seq0_bf + (size_t)t * Bp * H, nullptr,
                   h1_bf + (size_t)rp * Bp * H, Wih1_bf, Whh1_bf, bih1, bhh1,
                   h1_f, h1_bf + (size_t)wp * Bp * H,
                   enc_bf + (size_t)t * Bp * H, H, lds_gi);
    grid_barrier(cnt, gen);
  }
}

// ---- persistent decoder: 32 steps x 5 phases -----------------------------
__global__ void __launch_bounds__(NTHR) dec_kernel(
    const unsigned short* __restrict__ emb_bf,
    const unsigned short* __restrict__ wdih_bf, const unsigned short* __restrict__ wdhh_bf,
    const float* __restrict__ dbih, const float* __restrict__ dbhh,
    const unsigned short* __restrict__ enc_bf,
    const unsigned short* __restrict__ wattn_bf, const float* __restrict__ attn_b,
    const unsigned short* __restrict__ wfc_bf, const float* __restrict__ fc_b,
    unsigned short* __restrict__ h0_bf, float* __restrict__ h0_f,
    unsigned short* __restrict__ h1_bf, float* __restrict__ h1_f,
    unsigned short* __restrict__ comb_bf, unsigned short* __restrict__ hfc_bf,
    int* __restrict__ token, float* __restrict__ out, int* cnt, int* gen) {
  __shared__ float lds_gi[3][Bp][16];
  __shared__ float sScore[Sn];
  __shared__ float sRed[NTHR];
  const int w = threadIdx.x >> 5, lane = threadIdx.x & 31;
  const int coll = lane & 15, roff = (lane >> 4) << 3;
  const int tid = threadIdx.x;

  for (int t = 0; t < Tn; ++t) {
    const int rp = (t + 1) & 1, wp = t & 1;  // continue encoder ping-pong parity
    // Phase A: decoder GRU layer 0 (embedding gather input)
    gru_fused_step(emb_bf, token, h0_bf + (size_t)rp * Bp * H,
                   wdih_bf + 0, wdhh_bf + 0, dbih + 0, dbhh + 0,
                   h0_f, h0_bf + (size_t)wp * Bp * H, nullptr, 0, lds_gi);
    grid_barrier(cnt, gen);
    // Phase B: decoder GRU layer 1; out_t also goes into comb[:, 0:H]
    gru_fused_step(h0_bf + (size_t)wp * Bp * H, nullptr, h1_bf + (size_t)rp * Bp * H,
                   wdih_bf + (size_t)G * H, wdhh_bf + (size_t)G * H,
                   dbih + G, dbhh + G,
                   h1_f, h1_bf + (size_t)wp * Bp * H, comb_bf, 2 * H, lds_gi);
    grid_barrier(cnt, gen);
    // Phase C: attention scores + softmax + context (one block per batch row)
    if (blockIdx.x < Bv) {
      const int b = blockIdx.x;
      const float* qv = h1_f + (size_t)b * H;
      for (int s = tid; s < Sn; s += NTHR) {
        const unsigned short* er = enc_bf + (size_t)s * Bp * H + (size_t)b * H;
        float acc = 0.f;
        for (int h = 0; h < H; ++h) acc += bf2f(er[h]) * qv[h];
        sScore[s] = acc;
      }
      __syncthreads();
      float lm = -1e30f;
      for (int s = tid; s < Sn; s += NTHR) lm = fmaxf(lm, sScore[s]);
      sRed[tid] = lm; __syncthreads();
      for (int o = NTHR / 2; o > 0; o >>= 1) {
        if (tid < o) sRed[tid] = fmaxf(sRed[tid], sRed[tid + o]);
        __syncthreads();
      }
      const float mx = sRed[0]; __syncthreads();
      float ls = 0.f;
      for (int s = tid; s < Sn; s += NTHR) { float e = __expf(sScore[s] - mx); sScore[s] = e; ls += e; }
      sRed[tid] = ls; __syncthreads();
      for (int o = NTHR / 2; o > 0; o >>= 1) {
        if (tid < o) sRed[tid] += sRed[tid + o];
        __syncthreads();
      }
      const float inv = 1.0f / sRed[0]; __syncthreads();
      for (int s = tid; s < Sn; s += NTHR) {
        float wv = sScore[s] * inv; sScore[s] = wv;
        out[OATTN + (size_t)b * Sn * Tn + (size_t)s * Tn + t] = wv;
      }
      __syncthreads();
      for (int h = tid; h < H; h += NTHR) {
        float acc = 0.f;
        for (int s = 0; s < Sn; ++s)
          acc += sScore[s] * bf2f(enc_bf[(size_t)s * Bp * H + (size_t)b * H + h]);
        comb_bf[(size_t)b * 2 * H + H + h] = f2bf(acc);
      }
    }
    grid_barrier(cnt, gen);
    // Phase D: hfc = relu(comb @ attn_W^T + b), split-K across wave pairs
    {
      const int m = (w & 3) * 16, kh = (w >> 2) * 1024, n0 = blockIdx.x * 16;
      v8f acc = {};
      for (int k0 = 0; k0 < 1024; k0 += 32) {
        v16bf a = load_frag(comb_bf, 2 * H, m, kh + k0);
        v16bf bb = load_frag(wattn_bf, 2 * H, n0, kh + k0);
        acc = WMMA_BF16(a, bb, acc);
      }
      if (w >= 4) {
#pragma unroll
        for (int r = 0; r < 8; ++r) lds_gi[0][m + roff + r][coll] = acc[r];
      }
      __syncthreads();
      if (w < 4) {
        const int n = n0 + coll;
        const float bb = attn_b[n];
#pragma unroll
        for (int r = 0; r < 8; ++r) {
          int row = m + roff + r;
          float v = acc[r] + lds_gi[0][row][coll] + bb;
          hfc_bf[(size_t)row * H + n] = f2bf(v > 0.f ? v : 0.f);
        }
      }
    }
    grid_barrier(cnt, gen);
    // Phase E: logits + argmax feedback
    if (blockIdx.x < Bv) {
      const int b = blockIdx.x;
      if (tid < Vn) {
        float acc = fc_b[tid];
        const unsigned short* hr = hfc_bf + (size_t)b * H;
        const unsigned short* wr = wfc_bf + (size_t)tid * H;
        for (int h = 0; h < H; ++h) acc += bf2f(hr[h]) * bf2f(wr[h]);
        sRed[tid] = acc;
        out[(size_t)b * Tn * Vn + (size_t)t * Vn + tid] = acc;
      }
      __syncthreads();
      if (tid == 0) {
        int best = 0; float bvv = sRed[0];
        for (int v = 1; v < Vn; ++v) if (sRed[v] > bvv) { bvv = sRed[v]; best = v; }
        token[b] = best;
      }
    }
    grid_barrier(cnt, gen);
  }
  // final hidden states (L,B,H)
  if (blockIdx.x < Bv) {
    const int b = blockIdx.x;
    for (int h = tid; h < H; h += NTHR) {
      out[OHID + (size_t)b * H + h] = h0_f[(size_t)b * H + h];
      out[OHID + (size_t)Bv * H + (size_t)b * H + h] = h1_f[(size_t)b * H + h];
    }
  }
}

// ---- utility kernels -----------------------------------------------------
__global__ void cvt_bf16_kernel(const float* __restrict__ src, unsigned short* __restrict__ dst,
                                size_t n) {
  size_t i = (size_t)blockIdx.x * blockDim.x + threadIdx.x;
  size_t st = (size_t)gridDim.x * blockDim.x;
  for (; i < n; i += st) dst[i] = f2bf(src[i]);
}
__global__ void zero_kernel(unsigned int* __restrict__ p, size_t nwords) {
  size_t i = (size_t)blockIdx.x * blockDim.x + threadIdx.x;
  size_t st = (size_t)gridDim.x * blockDim.x;
  for (; i < nwords; i += st) p[i] = 0u;
}

extern "C" void kernel_launch(void* const* d_in, const int* in_sizes, int n_in,
                              void* d_out, int out_size, void* d_ws, size_t ws_size,
                              hipStream_t stream) {
  (void)in_sizes; (void)n_in; (void)out_size; (void)ws_size;
  const float* x    = (const float*)d_in[0];
  const float* emb  = (const float*)d_in[1];
  const float* Wih0 = (const float*)d_in[2];
  const float* Whh0 = (const float*)d_in[3];
  const float* bih0 = (const float*)d_in[4];
  const float* bhh0 = (const float*)d_in[5];
  const float* Wih1 = (const float*)d_in[6];
  const float* Whh1 = (const float*)d_in[7];
  const float* bih1 = (const float*)d_in[8];
  const float* bhh1 = (const float*)d_in[9];
  const float* dWih = (const float*)d_in[10];
  const float* dWhh = (const float*)d_in[11];
  const float* dbih = (const float*)d_in[12];
  const float* dbhh = (const float*)d_in[13];
  const float* attnW = (const float*)d_in[14];
  const float* attnb = (const float*)d_in[15];
  const float* fcW  = (const float*)d_in[16];
  const float* fcb  = (const float*)d_in[17];
  float* out = (float*)d_out;

  char* base = (char*)d_ws;
  size_t off = 0;
  auto alloc = [&](size_t bytes) -> void* {
    void* p = base + off;
    off += (bytes + 255) & ~(size_t)255;
    return p;
  };
  unsigned short* w_hh0 = (unsigned short*)alloc((size_t)G * H * 2);
  unsigned short* w_ih1 = (unsigned short*)alloc((size_t)G * H * 2);
  unsigned short* w_hh1 = (unsigned short*)alloc((size_t)G * H * 2);
  unsigned short* w_dih = (unsigned short*)alloc((size_t)2 * G * H * 2);
  unsigned short* w_dhh = (unsigned short*)alloc((size_t)2 * G * H * 2);
  unsigned short* w_attn = (unsigned short*)alloc((size_t)H * 2 * H * 2);
  unsigned short* w_fc  = (unsigned short*)alloc((size_t)Vn * H * 2);
  unsigned short* emb_bf = (unsigned short*)alloc((size_t)Vn * H * 2);
  unsigned short* seq0  = (unsigned short*)alloc((size_t)Sn * Bp * H * 2);
  unsigned short* encb  = (unsigned short*)alloc((size_t)Sn * Bp * H * 2);
  size_t state_beg = off;
  unsigned short* h0_bf = (unsigned short*)alloc((size_t)2 * Bp * H * 2);
  unsigned short* h1_bf = (unsigned short*)alloc((size_t)2 * Bp * H * 2);
  unsigned short* comb  = (unsigned short*)alloc((size_t)Bp * 2 * H * 2);
  unsigned short* hfc   = (unsigned short*)alloc((size_t)Bp * H * 2);
  float* h0_f = (float*)alloc((size_t)Bp * H * 4);
  float* h1_f = (float*)alloc((size_t)Bp * H * 4);
  int* token = (int*)alloc(64 * 4);
  int* cnt = (int*)alloc(4);
  int* gen = (int*)alloc(4);
  size_t state_end = off;

  // weight / embedding conversion to bf16 (once per call)
  cvt_bf16_kernel<<<512, 256, 0, stream>>>(Whh0, w_hh0, (size_t)G * H);
  cvt_bf16_kernel<<<512, 256, 0, stream>>>(Wih1, w_ih1, (size_t)G * H);
  cvt_bf16_kernel<<<512, 256, 0, stream>>>(Whh1, w_hh1, (size_t)G * H);
  cvt_bf16_kernel<<<512, 256, 0, stream>>>(dWih, w_dih, (size_t)2 * G * H);
  cvt_bf16_kernel<<<512, 256, 0, stream>>>(dWhh, w_dhh, (size_t)2 * G * H);
  cvt_bf16_kernel<<<512, 256, 0, stream>>>(attnW, w_attn, (size_t)H * 2 * H);
  cvt_bf16_kernel<<<64, 256, 0, stream>>>(fcW, w_fc, (size_t)Vn * H);
  cvt_bf16_kernel<<<64, 256, 0, stream>>>(emb, emb_bf, (size_t)Vn * H);
  // zero mutable state (hidden states, tokens, barrier counters)
  zero_kernel<<<256, 256, 0, stream>>>((unsigned int*)(base + state_beg),
                                       (state_end - state_beg) / 4);

  enc_l0_kernel<<<NBLK, NTHR, 0, stream>>>(x, Wih0, bih0, bhh0, w_hh0,
                                           h0_bf, h0_f, seq0, cnt, gen);
  enc_l1_kernel<<<NBLK, NTHR, 0, stream>>>(seq0, w_ih1, w_hh1, bih1, bhh1,
                                           h1_bf, h1_f, encb, cnt, gen);
  dec_kernel<<<NBLK, NTHR, 0, stream>>>(emb_bf, w_dih, w_dhh, dbih, dbhh, encb,
                                        w_attn, attnb, w_fc, fcb,
                                        h0_bf, h0_f, h1_bf, h1_f, comb, hfc,
                                        token, out, cnt, gen);
}
